// GCL_70351564309241
// MI455X (gfx1250) — matrix operand order repaired and checked by reading, not compiled, not run on hardware
//
#include <hip/hip_runtime.h>

typedef __attribute__((ext_vector_type(16))) _Float16 v16h;
typedef __attribute__((ext_vector_type(8)))  float    v8f;

#define DD    256     // input/output node feature dim
#define HH    256     // hidden dim
#define EDD   16      // edge attr dim
#define K1    528     // 2*DD + EDD
#define K1P   544     // K1 padded to multiple of 32
#define K1PS  552     // LDS row pad (bank-conflict-free)
#define KN    512     // DD + HH (node mlp input)
#define KNPS  520     // LDS row pad
#define HPS   264     // LDS row pad for 256-wide f16 rows
#define MPS   260     // LDS row pad for 256-wide f32 rows
#define EPB   64      // edges per block (4 WMMA M-tiles)
#define NORMI (1.0f/32.0f)

__device__ __forceinline__ float siluf(float x) { return x / (1.f + __expf(-x)); }
__device__ __forceinline__ float sigmf(float x) { return 1.f / (1.f + __expf(-x)); }

// Transpose-convert: src is [K][Ncols] f32 (row-major), dst is [Ncols][Kpad] f16,
// zero-padded for k >= K. Makes every WMMA B-fragment a contiguous 32B load.
__global__ void wcvt_kernel(const float* __restrict__ src, _Float16* __restrict__ dst,
                            int K, int Ncols, int Kpad) {
    int i = blockIdx.x * 256 + threadIdx.x;
    if (i >= Ncols * Kpad) return;
    int n = i / Kpad;
    int k = i - n * Kpad;
    dst[i] = (k < K) ? (_Float16)src[(size_t)k * Ncols + n] : (_Float16)0.f;
}

__global__ void zerof_kernel(float* __restrict__ p, int n) {
    int i = blockIdx.x * 256 + threadIdx.x;
    if (i < n) p[i] = 0.f;
}

// ---------------- edge model: 64 edges per block, 8 waves x (4 M-tiles x 2 N-tiles) ----------------
__global__ __launch_bounds__(256) void egcl_edge_kernel(
    const float* __restrict__ h, const int* __restrict__ ei,
    const float* __restrict__ eattr, const float* __restrict__ emask,
    const _Float16* __restrict__ We1t, const float* __restrict__ be1,
    const _Float16* __restrict__ We2t, const float* __restrict__ be2,
    const float* __restrict__ Wa, const float* __restrict__ ba,
    float* __restrict__ mij_out, float* __restrict__ agg, int Etot)
{
    // e_in tile (f16) overlaid with mij staging tile (f32) -- lifetimes are disjoint
    __shared__ __align__(16) unsigned char sAM[EPB * K1PS * 2];   // 70656 B >= 64*260*4
    _Float16 (*sA)[K1PS] = (_Float16(*)[K1PS])sAM;
    float    (*sM)[MPS]  = (float(*)[MPS])sAM;
    __shared__ _Float16 sX[EPB][HPS];
    __shared__ float    sPart[256];
    __shared__ float    sAtt[EPB];
    __shared__ int      sRow[EPB];
    __shared__ int      sCol[EPB];

    const int tid = threadIdx.x;
    const int e0  = blockIdx.x * EPB;

    if (tid < EPB)            sRow[tid]       = ei[e0 + tid];
    else if (tid < 2 * EPB)   sCol[tid - EPB] = ei[Etot + e0 + (tid - EPB)];
    __syncthreads();

    // gather e_in = [h[row] | h[col] | edge_attr | 0-pad] as f16
    for (int r = 0; r < EPB; ++r) {
        const int row = sRow[r], col = sCol[r];
        for (int k = tid; k < K1P; k += 256) {
            float v;
            if (k < DD)                 v = h[(size_t)row * DD + k];
            else if (k < 2 * DD)        v = h[(size_t)col * DD + (k - DD)];
            else if (k < 2 * DD + EDD)  v = eattr[(size_t)(e0 + r) * EDD + (k - 2 * DD)];
            else                        v = 0.f;
            sA[r][k] = (_Float16)v;
        }
    }
    __syncthreads();

    const int wave = tid >> 5;
    const int lane = tid & 31;
    const int m    = lane & 15;     // A row within M-tile / C column index
    const int kh   = lane >> 4;     // lane-half selects K group / M group
    const int n0   = wave * 32 + m; // column tile 2*wave
    const int n1   = n0 + 16;       // column tile 2*wave+1

    // ---- layer 1: [64 x 544] @ [544 x 256] ----
    v8f acc[4][2] = {};
    for (int kb = 0; kb < K1P; kb += 32) {
        v16h a[4];
        #pragma unroll
        for (int mt = 0; mt < 4; ++mt) {
            #pragma unroll
            for (int j = 0; j < 8; ++j) {
                const int k = kb + 2 * j + kh * 16;
                a[mt][2 * j]     = sA[mt * 16 + m][k];
                a[mt][2 * j + 1] = sA[mt * 16 + m][k + 1];
            }
        }
        const int ko = kb + kh * 16;
        v16h b0 = *(const v16h*)(We1t + (size_t)n0 * K1P + ko);
        v16h b1 = *(const v16h*)(We1t + (size_t)n1 * K1P + ko);
        #pragma unroll
        for (int mt = 0; mt < 4; ++mt) {
            acc[mt][0] = __builtin_amdgcn_wmma_f32_16x16x32_f16(false, a[mt], false, b0, (short)0, acc[mt][0], false, false);
            acc[mt][1] = __builtin_amdgcn_wmma_f32_16x16x32_f16(false, a[mt], false, b1, (short)0, acc[mt][1], false, false);
        }
    }
    #pragma unroll
    for (int mt = 0; mt < 4; ++mt) {
        #pragma unroll
        for (int r = 0; r < 8; ++r) {
            const int M = mt * 16 + r + kh * 8;
            sX[M][n0] = (_Float16)siluf(acc[mt][0][r] + be1[n0]);
            sX[M][n1] = (_Float16)siluf(acc[mt][1][r] + be1[n1]);
        }
    }
    __syncthreads();   // all waves done reading sA; sX ready

    // ---- layer 2: [64 x 256] @ [256 x 256] ----
    v8f dcc[4][2] = {};
    for (int kb = 0; kb < HH; kb += 32) {
        v16h a[4];
        #pragma unroll
        for (int mt = 0; mt < 4; ++mt) {
            #pragma unroll
            for (int j = 0; j < 8; ++j) {
                const int k = kb + 2 * j + kh * 16;
                a[mt][2 * j]     = sX[mt * 16 + m][k];
                a[mt][2 * j + 1] = sX[mt * 16 + m][k + 1];
            }
        }
        const int ko = kb + kh * 16;
        v16h b0 = *(const v16h*)(We2t + (size_t)n0 * HH + ko);
        v16h b1 = *(const v16h*)(We2t + (size_t)n1 * HH + ko);
        #pragma unroll
        for (int mt = 0; mt < 4; ++mt) {
            dcc[mt][0] = __builtin_amdgcn_wmma_f32_16x16x32_f16(false, a[mt], false, b0, (short)0, dcc[mt][0], false, false);
            dcc[mt][1] = __builtin_amdgcn_wmma_f32_16x16x32_f16(false, a[mt], false, b1, (short)0, dcc[mt][1], false, false);
        }
    }
    // mij epilogue: write to global + stage into sM (overlays sA, safe after sync above)
    #pragma unroll
    for (int mt = 0; mt < 4; ++mt) {
        #pragma unroll
        for (int r = 0; r < 8; ++r) {
            const int M = mt * 16 + r + kh * 8;
            float x0 = siluf(dcc[mt][0][r] + be2[n0]);
            float x1 = siluf(dcc[mt][1][r] + be2[n1]);
            sM[M][n0] = x0;
            sM[M][n1] = x1;
            mij_out[(size_t)(e0 + M) * HH + n0] = x0;   // mij output (pre-mask)
            mij_out[(size_t)(e0 + M) * HH + n1] = x1;
        }
    }
    __syncthreads();

    // ---- attention: att = sigmoid(mij @ Wa + ba) * edge_mask ----
    {
        const int e = tid >> 2, q = tid & 3;   // 4 lanes per edge
        float p = 0.f;
        #pragma unroll
        for (int j = 0; j < 64; ++j) p += sM[e][q * 64 + j] * Wa[q * 64 + j];
        sPart[tid] = p;
    }
    __syncthreads();
    if (tid < EPB) {
        float s = ba[0] + sPart[tid * 4] + sPart[tid * 4 + 1] + sPart[tid * 4 + 2] + sPart[tid * 4 + 3];
        sAtt[tid] = sigmf(s) * emask[e0 + tid];
    }
    __syncthreads();

    // ---- scatter-add edge_feat into agg (L2-resident atomics) ----
    for (int i = tid; i < EPB * HH; i += 256) {
        const int e = i >> 8, c = i & 255;
        atomicAdd(&agg[(size_t)sRow[e] * HH + c], sM[e][c] * sAtt[e]);
    }
}

// ---------------- node model: 16 nodes per block ----------------
__global__ __launch_bounds__(256) void egcl_node_kernel(
    const float* __restrict__ h, const float* __restrict__ agg,
    const float* __restrict__ nmask,
    const _Float16* __restrict__ Wn1t, const float* __restrict__ bn1,
    const _Float16* __restrict__ Wn2t, const float* __restrict__ bn2,
    float* __restrict__ hout)
{
    __shared__ _Float16 sA[16][KNPS];
    __shared__ _Float16 sX[16][HPS];

    const int tid = threadIdx.x;
    const int v0  = blockIdx.x * 16;

    for (int r = 0; r < 16; ++r) {
        const size_t node = v0 + r;
        for (int k = tid; k < KN; k += 256) {
            float v = (k < DD) ? h[node * DD + k] : agg[node * HH + (k - DD)] * NORMI;
            sA[r][k] = (_Float16)v;
        }
    }
    __syncthreads();

    const int wave = tid >> 5;
    const int lane = tid & 31;
    const int m    = lane & 15;
    const int kh   = lane >> 4;
    const int n0   = wave * 32 + m;
    const int n1   = n0 + 16;

    // ---- layer 1: [16 x 512] @ [512 x 256] ----
    v8f c0 = {}; v8f c1 = {};
    for (int kb = 0; kb < KN; kb += 32) {
        v16h a;
        #pragma unroll
        for (int j = 0; j < 8; ++j) {
            const int k = kb + 2 * j + kh * 16;
            a[2 * j]     = sA[m][k];
            a[2 * j + 1] = sA[m][k + 1];
        }
        const int ko = kb + kh * 16;
        v16h b0 = *(const v16h*)(Wn1t + (size_t)n0 * KN + ko);
        v16h b1 = *(const v16h*)(Wn1t + (size_t)n1 * KN + ko);
        c0 = __builtin_amdgcn_wmma_f32_16x16x32_f16(false, a, false, b0, (short)0, c0, false, false);
        c1 = __builtin_amdgcn_wmma_f32_16x16x32_f16(false, a, false, b1, (short)0, c1, false, false);
    }
    #pragma unroll
    for (int r = 0; r < 8; ++r) {
        const int M = r + kh * 8;
        sX[M][n0] = (_Float16)siluf(c0[r] + bn1[n0]);
        sX[M][n1] = (_Float16)siluf(c1[r] + bn1[n1]);
    }
    __syncthreads();

    // ---- layer 2 + residual + node_mask ----
    v8f d0 = {}; v8f d1 = {};
    for (int kb = 0; kb < HH; kb += 32) {
        v16h a;
        #pragma unroll
        for (int j = 0; j < 8; ++j) {
            const int k = kb + 2 * j + kh * 16;
            a[2 * j]     = sX[m][k];
            a[2 * j + 1] = sX[m][k + 1];
        }
        const int ko = kb + kh * 16;
        v16h b0 = *(const v16h*)(Wn2t + (size_t)n0 * HH + ko);
        v16h b1 = *(const v16h*)(Wn2t + (size_t)n1 * HH + ko);
        d0 = __builtin_amdgcn_wmma_f32_16x16x32_f16(false, a, false, b0, (short)0, d0, false, false);
        d1 = __builtin_amdgcn_wmma_f32_16x16x32_f16(false, a, false, b1, (short)0, d1, false, false);
    }
    #pragma unroll
    for (int r = 0; r < 8; ++r) {
        const int M = r + kh * 8;
        const size_t node = v0 + M;
        const float msk = nmask[node];
        hout[node * DD + n0] = (h[node * DD + n0] + d0[r] + bn2[n0]) * msk;
        hout[node * DD + n1] = (h[node * DD + n1] + d1[r] + bn2[n1]) * msk;
    }
}

// ---------------- workspace layout (bytes, 256B aligned) ----------------
#define WS_WE1T 0u         // 256*544 f16 = 278528 B
#define WS_WE2T 278528u    // 256*256 f16 = 131072 B
#define WS_WN1T 409600u    // 256*512 f16 = 262144 B
#define WS_WN2T 671744u    // 256*256 f16 = 131072 B
#define WS_AGG  802816u    // N*256  f32

extern "C" void kernel_launch(void* const* d_in, const int* in_sizes, int n_in,
                              void* d_out, int out_size, void* d_ws, size_t ws_size,
                              hipStream_t stream) {
    const float* h     = (const float*)d_in[0];
    const int*   ei    = (const int*)  d_in[1];
    const float* eattr = (const float*)d_in[2];
    const float* nmask = (const float*)d_in[3];
    const float* emask = (const float*)d_in[4];
    const float* We1   = (const float*)d_in[5];
    const float* be1   = (const float*)d_in[6];
    const float* We2   = (const float*)d_in[7];
    const float* be2   = (const float*)d_in[8];
    const float* Wa    = (const float*)d_in[9];
    const float* ba    = (const float*)d_in[10];
    const float* Wn1   = (const float*)d_in[11];
    const float* bn1   = (const float*)d_in[12];
    const float* Wn2   = (const float*)d_in[13];
    const float* bn2   = (const float*)d_in[14];

    const int Nn = in_sizes[3];  // node_mask: N elements
    const int Ee = in_sizes[4];  // edge_mask: E elements

    char* ws = (char*)d_ws;
    _Float16* We1t = (_Float16*)(ws + WS_WE1T);
    _Float16* We2t = (_Float16*)(ws + WS_WE2T);
    _Float16* Wn1t = (_Float16*)(ws + WS_WN1T);
    _Float16* Wn2t = (_Float16*)(ws + WS_WN2T);
    float*    agg  = (float*)   (ws + WS_AGG);

    float* hout    = (float*)d_out;                       // [N, 256]
    float* mij_out = (float*)d_out + (size_t)Nn * DD;     // [E, 256]

    // weight conversion (re-done every call: deterministic, capture-safe)
    wcvt_kernel<<<(HH * K1P + 255) / 256, 256, 0, stream>>>(We1, We1t, K1, HH, K1P);
    wcvt_kernel<<<(HH * HH  + 255) / 256, 256, 0, stream>>>(We2, We2t, HH, HH, HH);
    wcvt_kernel<<<(HH * KN  + 255) / 256, 256, 0, stream>>>(Wn1, Wn1t, KN, HH, KN);
    wcvt_kernel<<<(HH * HH  + 255) / 256, 256, 0, stream>>>(Wn2, Wn2t, HH, HH, HH);
    zerof_kernel<<<(Nn * HH + 255) / 256, 256, 0, stream>>>(agg, Nn * HH);

    egcl_edge_kernel<<<Ee / EPB, 256, 0, stream>>>(h, ei, eattr, emask,
                                                   We1t, be1, We2t, be2,
                                                   Wa, ba, mij_out, agg, Ee);
    egcl_node_kernel<<<Nn / 16, 256, 0, stream>>>(h, agg, nmask,
                                                  Wn1t, bn1, Wn2t, bn2, hout);
}
